// CausalSelfAttention_42666205118956
// MI455X (gfx1250) — compile-verified, matrix operand order checked by reading, not run on hardware
//
#include <hip/hip_runtime.h>

// ---------------------------------------------------------------- types
typedef __attribute__((ext_vector_type(16))) __bf16 v16bf;
typedef __attribute__((ext_vector_type(8)))  __bf16 v8bf;
typedef __attribute__((ext_vector_type(8)))  float  v8f;
typedef __attribute__((ext_vector_type(4)))  unsigned int u32x4;
typedef __attribute__((ext_vector_type(8)))  int i32x8;
typedef __attribute__((ext_vector_type(4)))  int i32x4;

union BF16x16 { v16bf v; v8bf h[2]; __bf16 e[16]; };
union F8      { v8f  v; float f[8]; };

#define Bq 4
#define Tq 2048
#define Dq 1024
#define Hq 16
#define HDq 64
#define Mrows (Bq*Tq)                 // 8192
#define QKVsz ((size_t)Bq*Hq*Tq*HDq)  // 8,388,608 elems per Q/K/V

#if defined(__has_builtin)
#if __has_builtin(__builtin_amdgcn_tensor_load_to_lds) && __has_builtin(__builtin_amdgcn_s_wait_tensorcnt)
#define HAVE_TDM 1
#endif
#endif
#ifndef HAVE_TDM
#define HAVE_TDM 0
#endif

static __device__ inline v8f zero8() {
    v8f z = {0.f,0.f,0.f,0.f,0.f,0.f,0.f,0.f};
    return z;
}

// Load a 16-element bf16 WMMA fragment row: elements 0-7 at K = half*8..+7,
// elements 8-15 at K = 16+half*8..+7 (A layout; B mirrored with N per lane).
static __device__ inline v16bf frag_ld(const __bf16* p, int half) {
    BF16x16 r;
    r.h[0] = *(const v8bf*)(p + half * 8);
    r.h[1] = *(const v8bf*)(p + 16 + half * 8);
    return r.v;
}

static __device__ inline v8f wmma_bf16(v16bf a, v16bf b, v8f c) {
    return __builtin_amdgcn_wmma_f32_16x16x32_bf16(
        /*neg_a=*/false, a, /*neg_b=*/false, b,
        /*c_mod=*/(short)0, c, /*reuse_a=*/false, /*reuse_b=*/false);
}

// intra-wave LDS fence: wait split DS counter + compiler barrier (CDNA5)
static __device__ inline void lds_fence() {
    asm volatile("s_wait_dscnt 0" ::: "memory");
}

#if HAVE_TDM
// Issue a TDM 2D tile load (tile0 x tile1 elems, 2-byte elements) into LDS.
// D# packing per CDNA5 ISA.
static __device__ inline void tdm_load_2d(unsigned int lds_off, const void* gaddr,
                                          unsigned int tdim0, unsigned int tdim1,
                                          unsigned int tile0, unsigned int tile1,
                                          unsigned int stride0) {
    unsigned long long ga = (unsigned long long)(uintptr_t)gaddr;
    u32x4 g0;
    g0[0] = 1u;                                    // count=1, user mode
    g0[1] = lds_off;                               // lds_addr
    g0[2] = (unsigned int)ga;                      // global_addr[31:0]
    g0[3] = (unsigned int)(ga >> 32) | (2u << 30); // global_addr[56:32] | type=2
    i32x8 g1;
    g1[0] = (int)(1u << 16);                       // data_size=1 (2 bytes)
    g1[1] = (int)((tdim0 & 0xffffu) << 16);        // tensor_dim0 lo16
    g1[2] = (int)((tdim0 >> 16) | ((tdim1 & 0xffffu) << 16)); // dim0 hi / dim1 lo
    g1[3] = (int)((tdim1 >> 16) | (tile0 << 16));  // dim1 hi | tile_dim0
    g1[4] = (int)tile1;                            // tile_dim1 (tile_dim2=0)
    g1[5] = (int)stride0;                          // tensor_dim0_stride lo32
    g1[6] = 0;                                     // stride0 hi16 | stride1 lo16
    g1[7] = 0;                                     // stride1 hi32
    i32x4 z4 = {0, 0, 0, 0};
#if __clang_major__ >= 23
    i32x8 z8 = {0, 0, 0, 0, 0, 0, 0, 0};
    __builtin_amdgcn_tensor_load_to_lds(g0, g1, z4, z4, z8, 0);
#else
    __builtin_amdgcn_tensor_load_to_lds(g0, g1, z4, z4, 0);
#endif
}
#endif

// ---------------------------------------------------------------- converts
__global__ __launch_bounds__(256) void k_f32_to_bf16(const float* __restrict__ src,
                                                     __bf16* __restrict__ dst, int n) {
    int i = blockIdx.x * 256 + threadIdx.x;
    if (i < n) dst[i] = (__bf16)src[i];
}

// src[R][C] fp32 -> dst[C][R] bf16 (K-major weights for B fragments)
__global__ __launch_bounds__(256) void k_transpose_bf16(const float* __restrict__ src,
                                                        __bf16* __restrict__ dst,
                                                        int R, int C) {
    int i = blockIdx.x * 256 + threadIdx.x;
    if (i < R * C) {
        int r = i / C, c = i % C;
        dst[(size_t)c * R + r] = (__bf16)src[i];
    }
}

// ---------------------------------------------------------------- GEMM
// C[M x NTOT] = A[M x 1024] * Bt[NTOT x 1024]^T + bias
// MODE 0: scatter Q,K -> [B,H,T,64] bf16, V -> [B,H,64,T] bf16 (pre-transposed
//         so attention's P*V B-fragments are contiguous loads).
// MODE 1: fp32 out row-major.
// LDS tiles are filled by the Tensor Data Mover (double-buffered) when available.
template<int NTOT, int MODE>
__global__ __launch_bounds__(256) void k_gemm(const __bf16* __restrict__ A,
                                              const __bf16* __restrict__ Bt,
                                              const float* __restrict__ bias,
                                              __bf16* __restrict__ qkv_out,
                                              float* __restrict__ f_out) {
    __shared__ alignas(16) __bf16 As[2][128 * 32];
    __shared__ alignas(16) __bf16 Bs[2][128 * 32];

    const int tid  = threadIdx.x;
    const int lane = tid & 31;
    const int w    = tid >> 5;
    const int half = lane >> 4;
    const int ln   = lane & 15;
    const int Mb   = blockIdx.y * 128;
    const int Nb   = blockIdx.x * 128;
    const int wm   = w >> 1;   // 0..3  (M sub-block of 32)
    const int wn   = w & 1;    // 0..1  (N sub-block of 64)

    v8f acc[2][4];
#pragma unroll
    for (int ms = 0; ms < 2; ++ms)
#pragma unroll
        for (int ns = 0; ns < 4; ++ns) acc[ms][ns] = zero8();

#if HAVE_TDM
    const __bf16* Atile = A  + (size_t)Mb * 1024;
    const __bf16* Btile = Bt + (size_t)Nb * 1024;
    if (w == 0) {
        tdm_load_2d((unsigned int)(uintptr_t)&As[0][0], Atile,      1024, Mrows, 32, 128, 1024);
        tdm_load_2d((unsigned int)(uintptr_t)&Bs[0][0], Btile,      1024, NTOT,  32, 128, 1024);
        tdm_load_2d((unsigned int)(uintptr_t)&As[1][0], Atile + 32, 1024, Mrows, 32, 128, 1024);
        tdm_load_2d((unsigned int)(uintptr_t)&Bs[1][0], Btile + 32, 1024, NTOT,  32, 128, 1024);
    }
    for (int i = 0; i < 32; ++i) {
        const int cur = i & 1;
        if (i < 31) __builtin_amdgcn_s_wait_tensorcnt(2);
        else        __builtin_amdgcn_s_wait_tensorcnt(0);
        __syncthreads();

        v16bf af[2], bf[4];
#pragma unroll
        for (int ms = 0; ms < 2; ++ms)
            af[ms] = frag_ld(&As[cur][(wm * 32 + ms * 16 + ln) * 32], half);
#pragma unroll
        for (int ns = 0; ns < 4; ++ns)
            bf[ns] = frag_ld(&Bs[cur][(wn * 64 + ns * 16 + ln) * 32], half);
#pragma unroll
        for (int ms = 0; ms < 2; ++ms)
#pragma unroll
            for (int ns = 0; ns < 4; ++ns)
                acc[ms][ns] = wmma_bf16(af[ms], bf[ns], acc[ms][ns]);

        __syncthreads();
        if (w == 0 && i + 2 < 32) {
            const int kb = (i + 2) * 32;
            tdm_load_2d((unsigned int)(uintptr_t)&As[cur][0], Atile + kb, 1024, Mrows, 32, 128, 1024);
            tdm_load_2d((unsigned int)(uintptr_t)&Bs[cur][0], Btile + kb, 1024, NTOT,  32, 128, 1024);
        }
    }
#else
    const int crow = tid >> 2;          // 0..63
    const int ccg  = (tid & 3) * 8;     // 0,8,16,24
    for (int kb = 0; kb < 1024; kb += 32) {
#pragma unroll
        for (int p = 0; p < 2; ++p) {
            int r = crow + p * 64;
            *(v8bf*)&As[0][r * 32 + ccg] =
                *(const v8bf*)&A[(size_t)(Mb + r) * 1024 + kb + ccg];
            *(v8bf*)&Bs[0][r * 32 + ccg] =
                *(const v8bf*)&Bt[(size_t)(Nb + r) * 1024 + kb + ccg];
        }
        __syncthreads();
        v16bf af[2], bf[4];
#pragma unroll
        for (int ms = 0; ms < 2; ++ms)
            af[ms] = frag_ld(&As[0][(wm * 32 + ms * 16 + ln) * 32], half);
#pragma unroll
        for (int ns = 0; ns < 4; ++ns)
            bf[ns] = frag_ld(&Bs[0][(wn * 64 + ns * 16 + ln) * 32], half);
#pragma unroll
        for (int ms = 0; ms < 2; ++ms)
#pragma unroll
            for (int ns = 0; ns < 4; ++ns)
                acc[ms][ns] = wmma_bf16(af[ms], bf[ns], acc[ms][ns]);
        __syncthreads();
    }
#endif

#pragma unroll
    for (int ms = 0; ms < 2; ++ms)
#pragma unroll
        for (int ns = 0; ns < 4; ++ns) {
            int n = Nb + wn * 64 + ns * 16 + ln;
            F8 c; c.v = acc[ms][ns];
#pragma unroll
            for (int r = 0; r < 8; ++r) {
                int m = Mb + wm * 32 + ms * 16 + r + 8 * half;
                float val = c.f[r] + bias[n];
                if (MODE == 0) {
                    int part = n >> 10, dd = n & 1023;
                    int hh = dd >> 6, hd = dd & 63;
                    int b = m >> 11, t = m & 2047;
                    size_t idx;
                    if (part == 2) // V transposed: [B,H,HD,T]
                        idx = (((size_t)b * Hq + hh) * HDq + hd) * Tq + t;
                    else           // Q,K: [B,H,T,HD]
                        idx = (((size_t)b * Hq + hh) * Tq + t) * HDq + hd;
                    qkv_out[(size_t)part * QKVsz + idx] = (__bf16)val;
                } else {
                    f_out[(size_t)m * NTOT + n] = val;
                }
            }
        }
}

// ---------------------------------------------------------------- flash attention
// grid (T/128, H, B), 256 threads; wave w owns 16 query rows.
// Q,K are [B,H,T,64]; VT is [B,H,64,T] (key-major) so all WMMA fragments are
// contiguous 128-bit loads; only P needs a (cheap) LDS layout hop.
// Key loop is split: mask-free main blocks + 1-2 masked diagonal blocks.
__global__ __launch_bounds__(256) void k_attn(const __bf16* __restrict__ Q,
                                              const __bf16* __restrict__ K,
                                              const __bf16* __restrict__ VT,
                                              __bf16* __restrict__ Y) {
    __shared__ alignas(16) __bf16 Plds[8][16 * 32];

    const int tid  = threadIdx.x;
    const int w    = tid >> 5;
    const int lane = tid & 31;
    const int half = lane >> 4;
    const int ln   = lane & 15;
    const int b    = blockIdx.z;
    const int h    = blockIdx.y;
    const int q0   = blockIdx.x * 128 + w * 16;

    const size_t hoff = ((size_t)b * Hq + h) * Tq * HDq;   // same for all three
    const __bf16* Qh  = Q  + hoff;
    const __bf16* Kh  = K  + hoff;
    const __bf16* VTh = VT + hoff;

    // Q fragments (A layout), fixed for the whole key loop
    v16bf qf[2];
#pragma unroll
    for (int s = 0; s < 2; ++s)
        qf[s] = frag_ld(Qh + (size_t)(q0 + ln) * 64 + s * 32, half);

    v8f accO[4];
#pragma unroll
    for (int nt = 0; nt < 4; ++nt) accO[nt] = zero8();
    float mrun[8], lrun[8];
#pragma unroll
    for (int r = 0; r < 8; ++r) { mrun[r] = -1.0e30f; lrun[r] = 0.f; }

    const float scale = 0.125f;            // 1/sqrt(64)
    const int qhi = q0 + 15;

    auto block = [&](int kbase, bool masked) {
        // prefetch next key block: K rows + VT rows
        if (kbase + 32 <= qhi) {
            __builtin_prefetch(Kh  + (size_t)(kbase + 32 + lane) * 64, 0, 1);
            __builtin_prefetch(VTh + (size_t)lane        * Tq + kbase + 32, 0, 1);
            __builtin_prefetch(VTh + (size_t)(lane + 32) * Tq + kbase + 32, 0, 1);
        }

        // K^T fragments (B layout): lane = key column; d contiguous in memory
        v16bf kf0[2], kf1[2];
#pragma unroll
        for (int s = 0; s < 2; ++s) {
            kf0[s] = frag_ld(Kh + (size_t)(kbase + ln) * 64 + s * 32, half);
            kf1[s] = frag_ld(Kh + (size_t)(kbase + 16 + ln) * 64 + s * 32, half);
        }
        v8f s0 = wmma_bf16(qf[0], kf0[0], zero8());
        s0     = wmma_bf16(qf[1], kf0[1], s0);
        v8f s1 = wmma_bf16(qf[0], kf1[0], zero8());
        s1     = wmma_bf16(qf[1], kf1[1], s1);
        F8 S0, S1; S0.v = s0; S1.v = s1;

        // V fragments (B layout): lane = d column; keys contiguous in VT rows
        v16bf vf[4];
#pragma unroll
        for (int nt = 0; nt < 4; ++nt)
            vf[nt] = frag_ld(VTh + (size_t)(nt * 16 + ln) * Tq + kbase, half);

        // online softmax (row M = r + 8*half, col N = lane&15 in C layout)
        const int key0 = kbase + ln;
        const int key1 = key0 + 16;
        float p0[8], p1[8], alpha[8];
#pragma unroll
        for (int r = 0; r < 8; ++r) {
            int q = q0 + r + 8 * half;
            float a = S0.f[r] * scale;
            float c = S1.f[r] * scale;
            if (masked) {
                if (key0 > q) a = -__builtin_inff();
                if (key1 > q) c = -__builtin_inff();
            }
            float mx = fmaxf(a, c);
#pragma unroll
            for (int off = 1; off < 16; off <<= 1)
                mx = fmaxf(mx, __shfl_xor(mx, off, 32));
            float mnew = fmaxf(mrun[r], mx);
            float al = __expf(mrun[r] - mnew);
            float e0 = __expf(a - mnew);
            float e1 = __expf(c - mnew);
            float rs = e0 + e1;
#pragma unroll
            for (int off = 1; off < 16; off <<= 1)
                rs += __shfl_xor(rs, off, 32);
            lrun[r] = lrun[r] * al + rs;
            mrun[r] = mnew;
            alpha[r] = al; p0[r] = e0; p1[r] = e1;
        }
#pragma unroll
        for (int nt = 0; nt < 4; ++nt) {
            F8 o; o.v = accO[nt];
#pragma unroll
            for (int r = 0; r < 8; ++r) o.f[r] *= alpha[r];
            accO[nt] = o.v;
        }

        // P: C layout -> A layout via per-wave LDS (only cross-lane hop left)
#pragma unroll
        for (int r = 0; r < 8; ++r) {
            int row = r + 8 * half;
            Plds[w][row * 32 + ln]      = (__bf16)p0[r];
            Plds[w][row * 32 + 16 + ln] = (__bf16)p1[r];
        }
        lds_fence();
        v16bf pf = frag_ld(&Plds[w][ln * 32], half);

#pragma unroll
        for (int nt = 0; nt < 4; ++nt)
            accO[nt] = wmma_bf16(pf, vf[nt], accO[nt]);
    };

    // mask-free main loop: all 32 keys <= q0 (min query row of this wave)
    const int nfull = (q0 >= 31) ? (((q0 - 31) >> 5) + 1) : 0;
    int kbase = 0;
    for (; kbase < nfull * 32; kbase += 32)
        block(kbase, false);
    // diagonal blocks (1-2) with causal masking
    for (; kbase <= qhi; kbase += 32)
        block(kbase, true);

    // normalize and store y[b][t][h*64+d] as bf16
#pragma unroll
    for (int nt = 0; nt < 4; ++nt) {
        F8 o; o.v = accO[nt];
#pragma unroll
        for (int r = 0; r < 8; ++r) {
            int q = q0 + r + 8 * half;
            int d = h * 64 + nt * 16 + ln;
            Y[((size_t)b * Tq + q) * Dq + d] = (__bf16)(o.f[r] / lrun[r]);
        }
    }
}

// ---------------------------------------------------------------- launch
extern "C" void kernel_launch(void* const* d_in, const int* in_sizes, int n_in,
                              void* d_out, int out_size, void* d_ws, size_t ws_size,
                              hipStream_t stream) {
    (void)in_sizes; (void)n_in; (void)out_size; (void)ws_size;

    const float* x      = (const float*)d_in[0];
    const float* w_attn = (const float*)d_in[1];
    const float* b_attn = (const float*)d_in[2];
    const float* w_proj = (const float*)d_in[3];
    const float* b_proj = (const float*)d_in[4];
    float* out = (float*)d_out;

    char* ws = (char*)d_ws;
    size_t o = 0;
    __bf16* xb    = (__bf16*)(ws + o); o += (size_t)Mrows * Dq * 2;        // 16 MB
    __bf16* wqkvT = (__bf16*)(ws + o); o += (size_t)3 * Dq * Dq * 2;       //  6 MB
    __bf16* wpT   = (__bf16*)(ws + o); o += (size_t)Dq * Dq * 2;           //  2 MB
    __bf16* Qb    = (__bf16*)(ws + o); o += 3 * QKVsz * 2;                 // 48 MB (Q,K,V^T)
    __bf16* yb    = (__bf16*)(ws + o); o += (size_t)Mrows * Dq * 2;        // 16 MB

    const int nx = Mrows * Dq;               // 8388608
    k_f32_to_bf16<<<(nx + 255) / 256, 256, 0, stream>>>(x, xb, nx);
    k_transpose_bf16<<<(3 * Dq * Dq + 255) / 256, 256, 0, stream>>>(w_attn, wqkvT, Dq, 3 * Dq);
    k_transpose_bf16<<<(Dq * Dq + 255) / 256, 256, 0, stream>>>(w_proj, wpT, Dq, Dq);

    // QKV: M=8192, N=3072 -> grid (24, 64)
    k_gemm<3072, 0><<<dim3(24, 64), 256, 0, stream>>>(xb, wqkvT, b_attn, Qb, nullptr);

    // attention: grid (T/128, H, B)
    k_attn<<<dim3(Tq / 128, Hq, Bq), 256, 0, stream>>>(Qb, Qb + QKVsz, Qb + 2 * QKVsz, yb);

    // projection: M=8192, N=1024 -> grid (8, 64)
    k_gemm<1024, 1><<<dim3(8, 64), 256, 0, stream>>>(yb, wpT, b_proj, nullptr, out);
}
